// DynamicGraphConv_79164837199987
// MI455X (gfx1250) — matrix operand order repaired and checked by reading, not compiled, not run on hardware
//
#include <hip/hip_runtime.h>
#include <stdint.h>

// Problem constants (match reference)
#define BB   16
#define NN   2048
#define DIN  256
#define DOUT 256
#define BN   (BB * NN)          // 32768 total rows

typedef __attribute__((ext_vector_type(16))) __bf16 v16bf;
typedef __attribute__((ext_vector_type(8)))  float  v8f;

// ---------- helpers ----------
static __device__ __forceinline__ unsigned short f2bf(float f) {
    unsigned u = __builtin_bit_cast(unsigned, f);
    unsigned r = u + 0x7FFFu + ((u >> 16) & 1u);   // round-to-nearest-even
    return (unsigned short)(r >> 16);
}
static __device__ __forceinline__ float bf2f(unsigned short h) {
    unsigned u = ((unsigned)h) << 16;
    return __builtin_bit_cast(float, u);
}

// Raw 32-bit LDS byte offset: per CDNA5 aperture rules the low 32 bits of a
// generic LDS address ARE the LDS offset (upper bits = SHARED_BASE aperture).
static __device__ __forceinline__ unsigned lds_raw(const void* p) {
    return (unsigned)(size_t)p;
}

// Async memory->LDS copy, 16B per lane, GVS addressing (SGPR base + VGPR u32
// offset). Tracked by ASYNCcnt; no VGPR data registers involved.
static __device__ __forceinline__ void async_load_b128(unsigned ldsaddr,
                                                       const void* gbase,
                                                       unsigned gofs) {
    asm volatile("global_load_async_to_lds_b128 %0, %1, %2"
                 :: "v"(ldsaddr), "v"(gofs), "s"(gbase)
                 : "memory");
}

#if defined(__has_builtin)
#if __has_builtin(__builtin_amdgcn_s_wait_asynccnt)
#define WAIT_ASYNC0() __builtin_amdgcn_s_wait_asynccnt(0)
#endif
#endif
#ifndef WAIT_ASYNC0
#define WAIT_ASYNC0() asm volatile("s_wait_asynccnt 0x0" ::: "memory")
#endif

// ---------- kernel 1: f32 -> bf16 conversion (4 elems/thread) ----------
__global__ void cvt_bf16_kernel(const float* __restrict__ src,
                                unsigned short* __restrict__ dst, int n4) {
    int t = blockIdx.x * 256 + threadIdx.x;
    if (t >= n4) return;
    float4 f = ((const float4*)src)[t];
    union { unsigned short us[4]; uint2 u; } pk;
    pk.us[0] = f2bf(f.x); pk.us[1] = f2bf(f.y);
    pk.us[2] = f2bf(f.z); pk.us[3] = f2bf(f.w);
    *(uint2*)(dst + (size_t)t * 4) = pk.u;
}

// ---------- kernel 2: h = x @ W^T, stored transposed as hT[b, o, n] (bf16) ----------
// One wave computes one 16x16 tile of h via 8 x v_wmma_f32_16x16x32_bf16.
__global__ void gemm_h_kernel(const unsigned short* __restrict__ xbf,
                              const unsigned short* __restrict__ Wbf,
                              unsigned short* __restrict__ hT) {
    const int wid   = blockIdx.x * 8 + (threadIdx.x >> 5);   // 32768 wave tiles
    const int lane  = threadIdx.x & 31;
    const int half  = lane >> 4;
    const int lan15 = lane & 15;

    const int iTile = wid >> 4;           // 2048 row tiles (16 rows each)
    const int oTile = wid & 15;           // 16 col tiles
    const int rBase = iTile * 16;
    const int oBase = oTile * 16;

    // A operand: row M = lane%16 of x; K interleaved by half-wave (ISA A layout)
    const unsigned short* xrow = xbf + (size_t)(rBase + lan15) * DIN;
    // B operand: col N = lane%16 maps to W row o; K interleaved by half-wave (same as A)
    const unsigned short* wrow = Wbf + (size_t)(oBase + lan15) * DIN;

    v8f c = {};
#pragma unroll
    for (int k0 = 0; k0 < DIN; k0 += 32) {
        union { uint4 q[2]; v16bf v; } Aop, Bop;
        Aop.q[0] = *(const uint4*)(xrow + k0 + half * 8);        // K k0+h8 .. +7
        Aop.q[1] = *(const uint4*)(xrow + k0 + 16 + half * 8);   // K k0+16+h8 ..
        Bop.q[0] = *(const uint4*)(wrow + k0 + half * 8);        // K k0+h8 .. +7
        Bop.q[1] = *(const uint4*)(wrow + k0 + 16 + half * 8);   // K k0+16+h8 ..
        c = __builtin_amdgcn_wmma_f32_16x16x32_bf16(
                false, Aop.v, false, Bop.v, (short)0, c, false, false);
    }

    // C/D layout: N = lane%16, rows M = half*8 + v  -> write 8 contiguous n's of hT
    const int b  = rBase >> 11;                       // N=2048 rows per batch
    const int n0 = (rBase & (NN - 1)) + half * 8;
    const int o  = oBase + lan15;
    union { unsigned short us[8]; uint4 q; } st;
#pragma unroll
    for (int v = 0; v < 8; ++v) st.us[v] = f2bf(c[v]);
    *(uint4*)(hT + ((size_t)(b * DOUT + o)) * NN + n0) = st.q;
}

// ---------- kernel 3: s1/s2 = h @ a1 / h @ a2 (from hT, coalesced over n) ----------
__global__ void score_kernel(const unsigned short* __restrict__ hT,
                             const float* __restrict__ a,
                             float* __restrict__ s1, float* __restrict__ s2) {
    int t = blockIdx.x * 256 + threadIdx.x;       // one (b,n) per thread
    if (t >= BN) return;
    int b = t >> 11, n = t & (NN - 1);
    const unsigned short* p = hT + (size_t)b * DOUT * NN + n;
    float a1s = 0.f, a2s = 0.f;
    for (int o = 0; o < DOUT; ++o) {
        float f = bf2f(p[(size_t)o * NN]);
        a1s += f * a[o];
        a2s += f * a[DOUT + o];
    }
    s1[t] = a1s; s2[t] = a2s;
}

// ---------- kernel 4: per-row softmax stats: m = max_j lrelu(si+sj), linv = 1/sum exp ----------
__global__ void stats_kernel(const float* __restrict__ s1, const float* __restrict__ s2,
                             float* __restrict__ m, float* __restrict__ linv) {
    int t = blockIdx.x * 256 + threadIdx.x;
    if (t >= BN) return;
    int b = t >> 11;
    float si = s1[t];
    const float* s2b = s2 + (size_t)b * NN;
    float mx = -3.402823466e38f;
    for (int j = 0; j < NN; ++j) {
        float e = si + s2b[j];
        e = (e > 0.f) ? e : 0.2f * e;
        mx = fmaxf(mx, e);
    }
    float sum = 0.f;
    for (int j = 0; j < NN; ++j) {
        float e = si + s2b[j];
        e = (e > 0.f) ? e : 0.2f * e;
        sum += __expf(e - mx);
    }
    m[t] = mx;
    linv[t] = 1.f / sum;
}

// ---------- kernel 5: out = softmax(E) @ h, flash-style, P built in registers ----------
// Block: 256 threads = 8 waves; 32 output rows x 256 cols per block.
// Waves 0-3: i-tile 0, waves 4-7: i-tile 1; each wave: 64 d-cols (4 accumulators).
// hT slabs (256 d x 64 j bf16, 32 KB each) are DOUBLE-BUFFERED in LDS and filled
// with GLOBAL_LOAD_ASYNC_TO_LDS (ASYNCcnt): slab k+1 is in flight underneath the
// WMMA loop on slab k. 72 KB static LDS total (WGP has 320 KB).
__global__ void attn_kernel(const unsigned short* __restrict__ hT,
                            const float* __restrict__ s1g, const float* __restrict__ s2g,
                            const float* __restrict__ mg,  const float* __restrict__ lg,
                            float* __restrict__ out) {
    __shared__ unsigned char ldsH[2][256 * 128];   // ping-pong: 256 d-rows x 64 j bf16
    __shared__ float ldsS2[NN];                    // s2 row for this batch (8 KB)

    const int tid   = threadIdx.x;
    const int wave  = tid >> 5;
    const int lane  = tid & 31;
    const int half  = lane >> 4;
    const int lan15 = lane & 15;

    const int r0    = blockIdx.x * 32;          // global row base (b*N + n)
    const int b     = r0 >> 11;
    const int iBase = r0 + (wave >> 2) * 16;    // this wave's 16-row tile
    const int dBase = (wave & 3) * 64;          // this wave's 64-col group

    // stage s2[b, :] into LDS once (visible after first barrier)
    for (int k = tid; k < NN; k += 256) ldsS2[k] = s2g[(size_t)b * NN + k];

    // per-lane row scalars (A-operand row M = lane%16 for both half-waves)
    const int   rowM = iBase + lan15;
    const float s1v  = s1g[rowM];
    const float mv   = mg[rowM];
    const float lv   = lg[rowM];                // 1/l — fold normalization into P

    v8f acc[4] = {v8f{}, v8f{}, v8f{}, v8f{}};
    const unsigned short* hTb = hT + (size_t)b * DOUT * NN;

    // this thread stages d-row `tid` of each slab: 128 B as 8 x async b128
    const unsigned ldsBase[2] = { lds_raw(&ldsH[0][0]) + (unsigned)tid * 128u,
                                  lds_raw(&ldsH[1][0]) + (unsigned)tid * 128u };
    const unsigned gRowOfs = (unsigned)tid * (NN * 2u);   // byte offset of d-row tid

    // prologue: slab 0 -> buffer 0
#pragma unroll
    for (int q = 0; q < 8; ++q)
        async_load_b128(ldsBase[0] + (unsigned)q * 16u, hTb,
                        gRowOfs + (unsigned)q * 16u);

    for (int it = 0; it < NN / 64; ++it) {
        const int j0 = it * 64;
        WAIT_ASYNC0();     // this wave's in-flight slab (buffer it&1) is in LDS
        __syncthreads();   // all waves: slab it visible; slab it-1 compute done

        // kick off slab it+1 into the other buffer while we compute slab it
        if (it + 1 < NN / 64) {
            const unsigned nb = ldsBase[(it + 1) & 1];
            const unsigned go = gRowOfs + (unsigned)(j0 + 64) * 2u;
#pragma unroll
            for (int q = 0; q < 8; ++q)
                async_load_b128(nb + (unsigned)q * 16u, hTb, go + (unsigned)q * 16u);
        }

        const unsigned char* cur = ldsH[it & 1];
#pragma unroll
        for (int kc = 0; kc < 2; ++kc) {
            // Build normalized P tile (16 x 32) in registers, ISA A-operand layout:
            // elems 0..7 -> K = half*8 + t ; elems 8..15 -> K = 16 + half*8 + (t-8)
            union { unsigned short us[16]; v16bf v; } A;
#pragma unroll
            for (int t = 0; t < 16; ++t) {
                int j = j0 + kc * 32 + half * 8 + ((t < 8) ? t : t + 8);
                float e = s1v + ldsS2[j];
                e = (e > 0.f) ? e : 0.2f * e;
                float p = __expf(e - mv) * lv;
                A.us[t] = f2bf(p);
            }
#pragma unroll
            for (int ac = 0; ac < 4; ++ac) {
                int d = dBase + ac * 16 + lan15;   // B-operand col N = lane%16
                union { uint4 q[2]; v16bf v; } Bop;
                const unsigned char* p = cur + d * 128 + kc * 64 + half * 16;
                Bop.q[0] = *(const uint4*)p; Bop.q[1] = *(const uint4*)(p + 32);
                acc[ac] = __builtin_amdgcn_wmma_f32_16x16x32_bf16(
                              false, A.v, false, Bop.v, (short)0, acc[ac], false, false);
            }
        }
    }

    // Writeback: C/D layout N = lane%16, M = half*8 + v
    const int nBase = (iBase & (NN - 1)) + half * 8;
#pragma unroll
    for (int ac = 0; ac < 4; ++ac) {
        int d = dBase + ac * 16 + lan15;
#pragma unroll
        for (int v = 0; v < 8; ++v) {
            out[((size_t)b * NN + (nBase + v)) * DOUT + d] = acc[ac][v];
        }
    }
}

// ---------- host launcher ----------
extern "C" void kernel_launch(void* const* d_in, const int* in_sizes, int n_in,
                              void* d_out, int out_size, void* d_ws, size_t ws_size,
                              hipStream_t stream) {
    const float* x = (const float*)d_in[0];   // (16, 2048, 256) f32
    const float* W = (const float*)d_in[1];   // (256, 256) f32
    const float* a = (const float*)d_in[2];   // (512,) f32
    float* out = (float*)d_out;               // (16, 2048, 256) f32

    // workspace carve-up
    uint8_t* ws = (uint8_t*)d_ws;
    size_t off = 0;
    unsigned short* xbf = (unsigned short*)(ws + off); off += (size_t)BN * DIN * 2;       // 16 MB
    unsigned short* Wbf = (unsigned short*)(ws + off); off += (size_t)DOUT * DIN * 2;     // 128 KB
    unsigned short* hT  = (unsigned short*)(ws + off); off += (size_t)BB * DOUT * NN * 2; // 16 MB
    float* s1   = (float*)(ws + off); off += (size_t)BN * 4;
    float* s2   = (float*)(ws + off); off += (size_t)BN * 4;
    float* mrow = (float*)(ws + off); off += (size_t)BN * 4;
    float* linv = (float*)(ws + off); off += (size_t)BN * 4;
    (void)ws_size; (void)in_sizes; (void)n_in; (void)out_size;

    // 1) f32 -> bf16
    {
        int n4 = (BN * DIN) / 4;
        cvt_bf16_kernel<<<(n4 + 255) / 256, 256, 0, stream>>>(x, xbf, n4);
        int w4 = (DOUT * DIN) / 4;
        cvt_bf16_kernel<<<(w4 + 255) / 256, 256, 0, stream>>>(W, Wbf, w4);
    }
    // 2) h GEMM (stores hT bf16): 32768 wave tiles / 8 waves per block
    gemm_h_kernel<<<(BN / 16) * (DOUT / 16) / 8, 256, 0, stream>>>(xbf, Wbf, hT);
    // 3) s1/s2
    score_kernel<<<BN / 256, 256, 0, stream>>>(hT, a, s1, s2);
    // 4) softmax stats
    stats_kernel<<<BN / 256, 256, 0, stream>>>(s1, s2, mrow, linv);
    // 5) attention matmul
    attn_kernel<<<BN / 32, 256, 0, stream>>>(hT, s1, s2, mrow, linv, out);
}
